// AR_GRU_decoder_82592221102338
// MI455X (gfx1250) — matrix operand order repaired and checked by reading, not compiled, not run on hardware
//
#include <hip/hip_runtime.h>
#include <hip/hip_bf16.h>
#include <cstddef>

// ---------------------------------------------------------------------------
// Autoregressive GRU decoder for MI455X (gfx1250, wave32, WMMA).
//
// Mapping: one wave32 per 16-batch tile (B=256 -> 16 blocks of 32 threads).
// Each wave runs the full T=4096 recurrence privately (no inter-block sync:
// batch rows are independent). Per step, three GEMMs via
// v_wmma_f32_16x16x32_bf16 with batch as M=16:
//   gi : [16 x 128] @ [128 x 96]   -> 6 N-tiles x 4 K-chunks = 24 WMMA
//   gh : [16 x  32] @ [ 32 x 96]   -> 6 WMMA (r/z accumulate into gi via C)
//   fc : [16 x  32] @ [ 32 x 16]   -> 1 WMMA (cols 0..4 valid)
// Weight B-fragments are packed into registers ONCE before the loop.
// ---------------------------------------------------------------------------

typedef __attribute__((ext_vector_type(16))) __bf16 v16bf;
typedef __attribute__((ext_vector_type(8)))  __bf16 v8bf;
typedef __attribute__((ext_vector_type(8)))  float  v8f;
typedef __attribute__((ext_vector_type(4)))  float  f4v;

#define IN_SZ 64
#define HID   32
#define NCLS  5

static __device__ inline v8f wmma_bf16(v16bf a, v16bf b, v8f c) {
  // (neg_a, A, neg_b, B, c_mod, C, reuse_a, reuse_b)
  return __builtin_amdgcn_wmma_f32_16x16x32_bf16(false, a, false, b,
                                                 (short)0, c, false, false);
}

static __device__ inline v8f splat8(float s) {
  v8f r;
#pragma unroll
  for (int i = 0; i < 8; ++i) r[i] = s;
  return r;
}

// Insert 8 consecutive bf16 values from LDS into fragment slots [e0..e0+7].
static __device__ inline void ld8_lds(v16bf& f, int e0, const __bf16* p) {
  v8bf v = *(const v8bf*)p;
#pragma unroll
  for (int i = 0; i < 8; ++i) f[e0 + i] = v[i];
}

// Insert 8 consecutive f32 values from global (cvt to bf16) into [e0..e0+7].
static __device__ inline void ld8_glb(v16bf& f, int e0, const float* p) {
  f4v a = *(const f4v*)p;
  f4v b = *(const f4v*)(p + 4);
#pragma unroll
  for (int i = 0; i < 4; ++i) {
    f[e0 + i]     = (__bf16)a[i];
    f[e0 + 4 + i] = (__bf16)b[i];
  }
}

__global__ __launch_bounds__(32, 1)
void ar_gru_decoder_kernel(const float* __restrict__ x,     // [B,T,64]
                           const float* __restrict__ w_ih,  // [96,128]
                           const float* __restrict__ w_hh,  // [96,32]
                           const float* __restrict__ b_ih,  // [96]
                           const float* __restrict__ b_hh,  // [96]
                           const float* __restrict__ fc_w,  // [5,32]
                           const float* __restrict__ fc_b,  // [5]
                           const float* __restrict__ emb,   // [5,64]
                           float* __restrict__ out,         // [B,T,5]
                           int T)
{
  // Row strides chosen so every 8-bf16 run starts 16B-aligned.
  __shared__ __align__(16) __bf16 prev_bf[16][72];  // prev embedding, bf16
  __shared__ __align__(16) __bf16 h_bf[16][48];     // hidden state,  bf16
  __shared__ __align__(16) float  lg_l[16][16];     // logits staging

  const int lane = threadIdx.x;
  const int m    = lane & 15;   // batch row within tile / D-col index
  const int half = lane >> 4;   // lane half (A/B fragment K-half select)
  const int b0   = blockIdx.x * 16;

  // ---- Pack weight B-fragments into registers (one time) -----------------
  // B layout (bf16 KxN=32x16): lane L -> col n = L&15, slot e -> K = (e<8?0:16) + 8*(L>>4) + (e&7).
  v16bf Bgi[6][4];  // w_ih^T tiles: [gate-tile][K-chunk]
  v16bf Bgh[6];     // w_hh^T tiles
  v16bf Bfc;        // fc_w^T tile (cols >= NCLS zeroed)
#pragma unroll
  for (int nt = 0; nt < 6; ++nt) {
    const int g = nt * 16 + m;  // gate row (= B column for this lane)
#pragma unroll
    for (int c = 0; c < 4; ++c)
#pragma unroll
      for (int e = 0; e < 16; ++e)
        Bgi[nt][c][e] = (__bf16)w_ih[g * 128 + c * 32 + (e < 8 ? 0 : 8) + half * 8 + e];
#pragma unroll
    for (int e = 0; e < 16; ++e)
      Bgh[nt][e] = (__bf16)w_hh[g * 32 + (e < 8 ? 0 : 8) + half * 8 + e];
  }
#pragma unroll
  for (int e = 0; e < 16; ++e)
    Bfc[e] = (m < NCLS) ? (__bf16)fc_w[m * 32 + (e < 8 ? 0 : 8) + half * 8 + e] : (__bf16)0.0f;

  // Per-lane biases (depend only on the D-column this lane owns).
  float brz[4], bin2[2], bhn2[2];
#pragma unroll
  for (int i = 0; i < 4; ++i) brz[i] = b_ih[i * 16 + m] + b_hh[i * 16 + m];
#pragma unroll
  for (int j = 0; j < 2; ++j) {
    bin2[j] = b_ih[64 + j * 16 + m];
    bhn2[j] = b_hh[64 + j * 16 + m];
  }
  const float bfc = (m < NCLS) ? fc_b[m] : 0.0f;

  // ---- Initial state: h = 0, prev = 0 ------------------------------------
  v8f hD[2];  // h_new in C/D layout (kept in registers across steps)
#pragma unroll
  for (int j = 0; j < 2; ++j) hD[j] = splat8(0.0f);
  v16bf Ah;   // h in A layout (for gh of next step & fc of this step)
#pragma unroll
  for (int e = 0; e < 16; ++e) Ah[e] = (__bf16)0.0f;
  {
    v8bf z;
#pragma unroll
    for (int i = 0; i < 8; ++i) z[i] = (__bf16)0.0f;
#pragma unroll
    for (int q = 0; q < 4; ++q)
      *(v8bf*)&prev_bf[m][half * 32 + q * 8] = z;
  }

  // ---- Time loop (sequential: argmax feedback) ----------------------------
  for (int t = 0; t < T; ++t) {
    const float* xrow = x + ((size_t)(b0 + m) * T + t) * IN_SZ;
    if (t + 1 < T) __builtin_prefetch(xrow + IN_SZ, 0, 1);  // global_prefetch

    // A fragments for gi (16x128 bf16): chunks 0,1 from x_t, 2,3 from prev.
    // A layout: lane L -> row m=L&15; slot e -> K = 32c + (e<8?0:16) + 8*half + (e&7).
    v16bf A[4];
#pragma unroll
    for (int c = 0; c < 2; ++c) {
      ld8_glb(A[c], 0, xrow + c * 32 + 8 * half);
      ld8_glb(A[c], 8, xrow + c * 32 + 16 + 8 * half);
    }
#pragma unroll
    for (int c = 0; c < 2; ++c) {
      ld8_lds(A[2 + c], 0, &prev_bf[m][c * 32 + 8 * half]);
      ld8_lds(A[2 + c], 8, &prev_bf[m][c * 32 + 16 + 8 * half]);
    }

    // Accumulators: tiles 0..3 = r,z gates (gi+gh merged via C operand);
    // tiles 4,5 = inn (gi only); hn kept separate for tanh(inn + r*hn).
    v8f acc[6], hn[2];
#pragma unroll
    for (int i = 0; i < 4; ++i) acc[i] = splat8(brz[i]);
#pragma unroll
    for (int j = 0; j < 2; ++j) {
      acc[4 + j] = splat8(bin2[j]);
      hn[j]      = splat8(bhn2[j]);
    }

#pragma unroll
    for (int nt = 0; nt < 6; ++nt)
#pragma unroll
      for (int c = 0; c < 4; ++c)
        acc[nt] = wmma_bf16(A[c], Bgi[nt][c], acc[nt]);   // 24 WMMA
#pragma unroll
    for (int nt = 0; nt < 4; ++nt)
      acc[nt] = wmma_bf16(Ah, Bgh[nt], acc[nt]);          // 4 WMMA
#pragma unroll
    for (int j = 0; j < 2; ++j)
      hn[j] = wmma_bf16(Ah, Bgh[4 + j], hn[j]);           // 2 WMMA

    // Gate math: fragment-local (gi/gh/h share C/D layout).
#pragma unroll
    for (int j = 0; j < 2; ++j) {
#pragma unroll
      for (int i = 0; i < 8; ++i) {
        float r  = 1.0f / (1.0f + __expf(-acc[j][i]));
        float zg = 1.0f / (1.0f + __expf(-acc[2 + j][i]));
        float nv = acc[4 + j][i] + r * hn[j][i];
        float e2 = __expf(-2.0f * nv);
        float th = (1.0f - e2) / (1.0f + e2);
        hD[j][i] = (1.0f - zg) * th + zg * hD[j][i];
      }
    }

    // h_new: D layout -> LDS (bf16), then gather back in A layout.
#pragma unroll
    for (int j = 0; j < 2; ++j)
#pragma unroll
      for (int v = 0; v < 8; ++v)
        h_bf[v + 8 * half][j * 16 + m] = (__bf16)hD[j][v];
    ld8_lds(Ah, 0, &h_bf[m][8 * half]);
    ld8_lds(Ah, 8, &h_bf[m][16 + 8 * half]);

    // fc: logits = h_new @ fc_w^T + fc_b  (1 WMMA, cols 0..4 valid)
    v8f lg = splat8(bfc);
    lg = wmma_bf16(Ah, Bfc, lg);
#pragma unroll
    for (int v = 0; v < 8; ++v) lg_l[v + 8 * half][m] = lg[v];

    // Per-row log_softmax + argmax (both lane-halves mirror row m so EXEC
    // stays full for the next iteration's WMMAs).
    float l[NCLS];
#pragma unroll
    for (int c = 0; c < NCLS; ++c) l[c] = lg_l[m][c];
    float mx = l[0];
    int  idx = 0;
#pragma unroll
    for (int c = 1; c < NCLS; ++c)
      if (l[c] > mx) { mx = l[c]; idx = c; }   // first-max, matches argmax
    float se = 0.0f;
#pragma unroll
    for (int c = 0; c < NCLS; ++c) se += __expf(l[c] - mx);
    const float lse = mx + __logf(se);

    if (half == 0) {
      float* orow = out + ((size_t)(b0 + m) * T + t) * NCLS;
#pragma unroll
      for (int c = 0; c < NCLS; ++c) orow[c] = l[c] - lse;
    }

    // prev = emb[idx]  (each half-lane writes half of its row, bf16)
    const float* er = emb + idx * IN_SZ + half * 32;
#pragma unroll
    for (int q = 0; q < 4; ++q) {
      f4v a = *(const f4v*)(er + q * 8);
      f4v b = *(const f4v*)(er + q * 8 + 4);
      v8bf pv;
#pragma unroll
      for (int i = 0; i < 4; ++i) {
        pv[i]     = (__bf16)a[i];
        pv[4 + i] = (__bf16)b[i];
      }
      *(v8bf*)&prev_bf[m][half * 32 + q * 8] = pv;
    }
  }
}

extern "C" void kernel_launch(void* const* d_in, const int* in_sizes, int n_in,
                              void* d_out, int out_size, void* d_ws, size_t ws_size,
                              hipStream_t stream) {
  const float* x    = (const float*)d_in[0];
  const float* w_ih = (const float*)d_in[1];
  const float* w_hh = (const float*)d_in[2];
  const float* b_ih = (const float*)d_in[3];
  const float* b_hh = (const float*)d_in[4];
  const float* fc_w = (const float*)d_in[5];
  const float* fc_b = (const float*)d_in[6];
  const float* emb  = (const float*)d_in[7];
  // d_in[8] = x_lengths (unused by the reference computation)

  const int B = in_sizes[8];                 // x_lengths has B elements
  const int T = in_sizes[0] / (B * IN_SZ);   // x is [B,T,64]

  dim3 grid(B / 16), block(32);
  ar_gru_decoder_kernel<<<grid, block, 0, stream>>>(
      x, w_ih, w_hh, b_ih, b_hh, fc_w, fc_b, emb, (float*)d_out, T);
}
